// PointNet_4810363372407
// MI455X (gfx1250) — compile-verified, matrix operand order checked by reading, not run on hardware
//
#include <hip/hip_runtime.h>

// ---------------------------------------------------------------------------
// PointNet-conv GNN on gfx1250: bf16 WMMA GEMMs (32x64 per wave) +
// fp32 atomic scatter-mean.
// ---------------------------------------------------------------------------

typedef __bf16 bf16_t;
typedef bf16_t v16bf __attribute__((ext_vector_type(16)));
typedef float  v8f   __attribute__((ext_vector_type(8)));

#define NN 20000          // nodes (multiple of 32)
#define NE 320000         // edges (multiple of 32)
#define INCH 16
#define CHUNK_E 16384     // edge chunk (mult of 32; remainder 8704 = 32*272)
#define GEMM_WAVES 4      // waves per GEMM block (block = 128 threads)
#define NBLK 4            // 16-wide N tiles per wave
#define MBLK 2            // 16-row M tiles per wave  (wave computes 32x64)

// ---- bf16 helpers (round-to-nearest-even), storage as u16 ------------------
__device__ __forceinline__ unsigned short f2bf(float f) {
  unsigned int u = __float_as_uint(f);
  u += 0x7FFFu + ((u >> 16) & 1u);
  return (unsigned short)(u >> 16);
}

// ---------------------------------------------------------------------------
// Small utility kernels
// ---------------------------------------------------------------------------
__global__ void k_zero_f32(float* p, int n) {
  int i = blockIdx.x * blockDim.x + threadIdx.x;
  if (i < n) p[i] = 0.0f;
}

__global__ void k_count(const long long* dst, float* cnt, int nE) {
  int e = blockIdx.x * blockDim.x + threadIdx.x;
  if (e < nE) atomicAdd(&cnt[(int)dst[e]], 1.0f);
}

__global__ void k_invcnt(float* c, int n) {
  int i = blockIdx.x * blockDim.x + threadIdx.x;
  if (i < n) c[i] = 1.0f / fmaxf(c[i], 1.0f);
}

__global__ void k_f32_to_bf(const float* x, unsigned short* y, int n) {
  int i = blockIdx.x * blockDim.x + threadIdx.x;
  if (i < n) y[i] = f2bf(x[i]);
}

// w [K x N] fp32 -> wT [N x Kpad] bf16 (zero-padded along K)
__global__ void k_wT_bf(const float* w, unsigned short* wT, int K, int N, int Kpad) {
  int k = blockIdx.x * blockDim.x + threadIdx.x;
  int n = blockIdx.y;
  if (k >= Kpad) return;
  wT[(size_t)n * Kpad + k] = (k < K) ? f2bf(w[(size_t)k * N + n]) : (unsigned short)0;
}

// Build edge features: ef[e][:] = [ h_bf[src[e]][0..C-1], pos_src - pos_dst, 0... ]
__global__ void k_gather(const unsigned short* hb, int C, const float* pos,
                         const long long* src, const long long* dst,
                         unsigned short* ef, int Kpad, int nE) {
  int k = blockIdx.x * blockDim.x + threadIdx.x;
  int e = blockIdx.y;
  if (k >= Kpad || e >= nE) return;
  unsigned short v;
  if (k < C) {
    int s = (int)src[e];
    v = hb[(size_t)s * C + k];
  } else if (k == C) {
    int s = (int)src[e]; int d = (int)dst[e];
    v = f2bf(pos[2 * s] - pos[2 * d]);
  } else if (k == C + 1) {
    int s = (int)src[e]; int d = (int)dst[e];
    v = f2bf(pos[2 * s + 1] - pos[2 * d + 1]);
  } else {
    v = 0;
  }
  ef[(size_t)e * Kpad + k] = v;
}

// h_bf[n][c] = bf16( relu( agg[n][c] * inv[n] ) )
__global__ void k_finalize(const float* agg, const float* inv, unsigned short* hb, int H) {
  int c = blockIdx.x * blockDim.x + threadIdx.x;
  int n = blockIdx.y;
  if (c < H) hb[(size_t)n * H + c] = f2bf(fmaxf(agg[(size_t)n * H + c] * inv[n], 0.0f));
}

// final tiny head layer: out[n][a] = dot(x[n][:], w[:,a]) + b[a]   (K=512, A=4)
__global__ void k_head3(const float* x, const float* w, const float* b,
                        float* out, int K, int NA, int M) {
  int t = blockIdx.x * blockDim.x + threadIdx.x;
  if (t >= M * NA) return;
  int n = t / NA, a = t % NA;
  const float* xr = x + (size_t)n * K;
  float s = b[a];
  for (int k = 0; k < K; ++k) s += xr[k] * w[(size_t)k * NA + a];
  out[t] = s;
}

// ---------------------------------------------------------------------------
// WMMA fragment loader: 16 bf16 per lane, ISA 16-bit A-layout
//   elems 0..7  at  k + 8*half ;  elems 8..15 at  k + 16 + 8*half
// ---------------------------------------------------------------------------
__device__ __forceinline__ v16bf load_frag(const unsigned short* row, int k, int half) {
  union { uint4 q[2]; v16bf v; } f;
  f.q[0] = *(const uint4*)(row + k + half * 8);
  f.q[1] = *(const uint4*)(row + k + 16 + half * 8);
  return f.v;
}

// ---------------------------------------------------------------------------
// WMMA bf16 GEMM:  D = A[M x K] * B^T_rows[N x K] (+bias) (relu?) -> f32 or bf16
// Wave computes a (16*MBLK) x (16*NBLK) = 32x64 block:
//   per 32-wide K-step: 4 A-loads + 8 B-loads feed 8 WMMAs (1.5 loads/WMMA).
// M is a multiple of 32 and all N are multiples of 64 -> guards wave-uniform,
// EXEC stays all-ones for WMMA.
// ---------------------------------------------------------------------------
__global__ void k_wmma_gemm(const unsigned short* A, int lda,
                            const unsigned short* BT, int ldb,
                            const float* bias,
                            float* outF, unsigned short* outB, int ldo,
                            int N, int K, int relu) {
  int wave = threadIdx.x >> 5;
  int lane = threadIdx.x & 31;
  int mt0 = blockIdx.x * MBLK;
  int nt0 = (blockIdx.y * GEMM_WAVES + wave) * NBLK;
  if (nt0 * 16 >= N) return;           // wave-uniform
  int mr = lane & 15, half = lane >> 4;
  const unsigned short* arow = A + (size_t)(mt0 * 16 + mr) * lda;
  const unsigned short* brow = BT + (size_t)(nt0 * 16 + mr) * ldb;

  v8f acc[MBLK][NBLK];
#pragma unroll
  for (int i = 0; i < MBLK; ++i)
#pragma unroll
    for (int j = 0; j < NBLK; ++j)
      acc[i][j] = (v8f){0.f,0.f,0.f,0.f,0.f,0.f,0.f,0.f};

  for (int k = 0; k < K; k += 32) {
    __builtin_prefetch(arow + k + 256, 0, 1);   // global_prefetch_b8, A stream
    v16bf fa[MBLK];
#pragma unroll
    for (int i = 0; i < MBLK; ++i)
      fa[i] = load_frag(arow + (size_t)i * 16 * lda, k, half);
#pragma unroll
    for (int j = 0; j < NBLK; ++j) {
      v16bf fb = load_frag(brow + (size_t)j * 16 * ldb, k, half);
#pragma unroll
      for (int i = 0; i < MBLK; ++i)
        acc[i][j] = __builtin_amdgcn_wmma_f32_16x16x32_bf16(
            false, fa[i], false, fb, (short)0, acc[i][j], false, false);
    }
  }

  int n = lane & 15;
#pragma unroll
  for (int i = 0; i < MBLK; ++i) {
#pragma unroll
    for (int j = 0; j < NBLK; ++j) {
      int ng = (nt0 + j) * 16 + n;
      float bv = bias ? bias[ng] : 0.0f;
#pragma unroll
      for (int r = 0; r < 8; ++r) {
        int mg = (mt0 + i) * 16 + r + 8 * half;
        float v = acc[i][j][r] + bv;
        if (relu) v = fmaxf(v, 0.0f);
        if (outF) outF[(size_t)mg * ldo + ng] = v;
        else      outB[(size_t)mg * ldo + ng] = f2bf(v);
      }
    }
  }
}

// GEMM + fused per-edge bias + atomic scatter-add into agg[dst[e]][:]
__global__ void k_wmma_scatter(const unsigned short* A, int lda,
                               const unsigned short* BT, int ldb,
                               const float* bias,
                               const long long* dst, float* agg,
                               int N, int K) {
  int wave = threadIdx.x >> 5;
  int lane = threadIdx.x & 31;
  int mt0 = blockIdx.x * MBLK;
  int nt0 = (blockIdx.y * GEMM_WAVES + wave) * NBLK;
  if (nt0 * 16 >= N) return;           // wave-uniform
  int mr = lane & 15, half = lane >> 4;
  const unsigned short* arow = A + (size_t)(mt0 * 16 + mr) * lda;
  const unsigned short* brow = BT + (size_t)(nt0 * 16 + mr) * ldb;

  v8f acc[MBLK][NBLK];
#pragma unroll
  for (int i = 0; i < MBLK; ++i)
#pragma unroll
    for (int j = 0; j < NBLK; ++j)
      acc[i][j] = (v8f){0.f,0.f,0.f,0.f,0.f,0.f,0.f,0.f};

  for (int k = 0; k < K; k += 32) {
    __builtin_prefetch(arow + k + 256, 0, 1);
    v16bf fa[MBLK];
#pragma unroll
    for (int i = 0; i < MBLK; ++i)
      fa[i] = load_frag(arow + (size_t)i * 16 * lda, k, half);
#pragma unroll
    for (int j = 0; j < NBLK; ++j) {
      v16bf fb = load_frag(brow + (size_t)j * 16 * ldb, k, half);
#pragma unroll
      for (int i = 0; i < MBLK; ++i)
        acc[i][j] = __builtin_amdgcn_wmma_f32_16x16x32_bf16(
            false, fa[i], false, fb, (short)0, acc[i][j], false, false);
    }
  }

  int n = lane & 15;
#pragma unroll
  for (int i = 0; i < MBLK; ++i) {
#pragma unroll
    for (int r = 0; r < 8; ++r) {
      int eg = (mt0 + i) * 16 + r + 8 * half;
      int node = (int)dst[eg];
      float* aggrow = agg + (size_t)node * N;
#pragma unroll
      for (int j = 0; j < NBLK; ++j) {
        int ng = (nt0 + j) * 16 + n;
        atomicAdd(&aggrow[ng], acc[i][j][r] + bias[ng]);
      }
    }
  }
}

// ---------------------------------------------------------------------------
// Host orchestration
// ---------------------------------------------------------------------------
extern "C" void kernel_launch(void* const* d_in, const int* in_sizes, int n_in,
                              void* d_out, int out_size, void* d_ws, size_t ws_size,
                              hipStream_t stream) {
  (void)in_sizes; (void)n_in; (void)out_size; (void)ws_size;

  const float*     h0   = (const float*)d_in[0];
  const float*     pos  = (const float*)d_in[1];
  const long long* ei   = (const long long*)d_in[2];
  const long long* srcE = ei;
  const long long* dstE = ei + NE;

  // layer params: l{1..4}_{w1,b1,w2,b2} at indices 3..18, head at 19..24
  const float* LW1[4] = {(const float*)d_in[3],  (const float*)d_in[7],
                         (const float*)d_in[11], (const float*)d_in[15]};
  const float* LB1[4] = {(const float*)d_in[4],  (const float*)d_in[8],
                         (const float*)d_in[12], (const float*)d_in[16]};
  const float* LW2[4] = {(const float*)d_in[5],  (const float*)d_in[9],
                         (const float*)d_in[13], (const float*)d_in[17]};
  const float* LB2[4] = {(const float*)d_in[6],  (const float*)d_in[10],
                         (const float*)d_in[14], (const float*)d_in[18]};
  const float* hd_w1 = (const float*)d_in[19];
  const float* hd_b1 = (const float*)d_in[20];
  const float* hd_w2 = (const float*)d_in[21];
  const float* hd_b2 = (const float*)d_in[22];
  const float* hd_w3 = (const float*)d_in[23];
  const float* hd_b3 = (const float*)d_in[24];

  const int Cin[4]  = {INCH, 128, 256, 1024};
  const int H1[4]   = {128, 256, 1024, 1024};
  const int H2[4]   = {128, 256, 1024, 1024};
  const int Kpad[4] = {32, 160, 288, 1056};   // round32(Cin+2)

  const int NT = 16 * NBLK * GEMM_WAVES;      // cols covered per block = 256
  const int MT = 16 * MBLK;                   // rows covered per block = 32

  // ---- carve workspace ----
  char* wsp = (char*)d_ws;
  auto carve = [&](size_t bytes) -> void* {
    void* p = (void*)wsp;
    wsp += (bytes + 255) & ~(size_t)255;
    return p;
  };
  float*          inv  = (float*)carve((size_t)NN * 4);
  unsigned short* hb   = (unsigned short*)carve((size_t)NN * 1024 * 2);
  unsigned short* t1   = (unsigned short*)carve((size_t)NN * 1024 * 2);
  float*          agg  = (float*)carve((size_t)NN * 1024 * 4);
  unsigned short* ef   = (unsigned short*)carve((size_t)CHUNK_E * 1056 * 2);
  unsigned short* msg  = (unsigned short*)carve((size_t)CHUNK_E * 1024 * 2);

  unsigned short* WB1[4]; unsigned short* WB2[4];
  for (int l = 0; l < 4; ++l) {
    WB1[l] = (unsigned short*)carve((size_t)H1[l] * Kpad[l] * 2); // [N x Kpad]
    WB2[l] = (unsigned short*)carve((size_t)H2[l] * H1[l] * 2);   // [N x K]
  }
  unsigned short* WBH1 = (unsigned short*)carve((size_t)1024 * 1024 * 2);
  unsigned short* WBH2 = (unsigned short*)carve((size_t)512 * 1024 * 2);

  // ---- one-time conversions ----
  for (int l = 0; l < 4; ++l) {
    k_wT_bf<<<dim3((Kpad[l] + 255) / 256, H1[l]), 256, 0, stream>>>(
        LW1[l], WB1[l], Cin[l] + 2, H1[l], Kpad[l]);
    k_wT_bf<<<dim3((H1[l] + 255) / 256, H2[l]), 256, 0, stream>>>(
        LW2[l], WB2[l], H1[l], H2[l], H1[l]);
  }
  k_wT_bf<<<dim3(4, 1024), 256, 0, stream>>>(hd_w1, WBH1, 1024, 1024, 1024);
  k_wT_bf<<<dim3(4, 512),  256, 0, stream>>>(hd_w2, WBH2, 1024,  512, 1024);

  k_f32_to_bf<<<(NN * INCH + 255) / 256, 256, 0, stream>>>(h0, hb, NN * INCH);

  // in-degree -> inv_cnt
  k_zero_f32<<<(NN + 255) / 256, 256, 0, stream>>>(inv, NN);
  k_count<<<(NE + 255) / 256, 256, 0, stream>>>(dstE, inv, NE);
  k_invcnt<<<(NN + 255) / 256, 256, 0, stream>>>(inv, NN);

  // ---- 4 message-passing layers ----
  for (int l = 0; l < 4; ++l) {
    const int kp = Kpad[l], c = Cin[l], h1 = H1[l], h2 = H2[l];
    k_zero_f32<<<((NN * h2) + 255) / 256, 256, 0, stream>>>(agg, NN * h2);

    for (int base = 0; base < NE; base += CHUNK_E) {
      int nE = NE - base; if (nE > CHUNK_E) nE = CHUNK_E;
      // gather edge features
      k_gather<<<dim3((kp + 63) / 64, nE), 64, 0, stream>>>(
          hb, c, pos, srcE + base, dstE + base, ef, kp, nE);
      // msg = relu(ef @ w1 + b1)  -> bf16
      k_wmma_gemm<<<dim3(nE / MT, (h1 + NT - 1) / NT), 32 * GEMM_WAVES, 0, stream>>>(
          ef, kp, WB1[l], kp, LB1[l], (float*)nullptr, msg, h1, h1, kp, 1);
      // agg[dst] += msg @ w2 + b2
      k_wmma_scatter<<<dim3(nE / MT, (h2 + NT - 1) / NT), 32 * GEMM_WAVES, 0, stream>>>(
          msg, h1, WB2[l], h1, LB2[l], dstE + base, agg, h2, h1);
    }
    // h = relu(agg * inv_cnt) -> bf16
    k_finalize<<<dim3((h2 + 255) / 256, NN), 256, 0, stream>>>(agg, inv, hb, h2);
  }

  // ---- head MLP ----
  // t1 = relu(h @ hd_w1 + b1)  (bf16)
  k_wmma_gemm<<<dim3(NN / MT, (1024 + NT - 1) / NT), 32 * GEMM_WAVES, 0, stream>>>(
      hb, 1024, WBH1, 1024, hd_b1, (float*)nullptr, t1, 1024, 1024, 1024, 1);
  // t2 = relu(t1 @ hd_w2 + b2) (fp32, reuse agg buffer, stride 512)
  k_wmma_gemm<<<dim3(NN / MT, (512 + NT - 1) / NT), 32 * GEMM_WAVES, 0, stream>>>(
      t1, 1024, WBH2, 1024, hd_b2, agg, (unsigned short*)nullptr, 512, 512, 1024, 1);
  // out = t2 @ hd_w3 + b3
  k_head3<<<(NN * 4 + 255) / 256, 256, 0, stream>>>(
      agg, hd_w3, hd_b3, (float*)d_out, 512, 4, NN);
}
